// TransMIL_45664092291181
// MI455X (gfx1250) — matrix operand order repaired and checked by reading, not compiled, not run on hardware
//
#include <hip/hip_runtime.h>
#include <hip/hip_bf16.h>
#include <math.h>

// ---------------------------------------------------------------------------
// TransMIL-style transformer forward on MI455X (gfx1250).
// bf16 WMMA everywhere; GEMMs feed the matrix pipes through the Tensor Data
// Mover (TDM -> LDS -> ds_load_b128 -> v_wmma), double-buffered per 32-K step.
// ---------------------------------------------------------------------------

typedef __attribute__((ext_vector_type(16))) __bf16 v16bf;
typedef __attribute__((ext_vector_type(8)))  float  v8f;
typedef unsigned short u16b;

union BFrag { v16bf v; uint4 q[2]; u16b us[16]; };

__device__ __forceinline__ u16b f2bf(float f) {
  unsigned int u = __float_as_uint(f);
  unsigned int r = (u + 0x7FFFu + ((u >> 16) & 1u)) >> 16;   // RNE
  return (u16b)r;
}
__device__ __forceinline__ float geluf(float x) {
  return 0.5f * x * (1.0f + erff(x * 0.70710678118654752f)); // exact erf gelu
}

// Model constants
#define BATCH 4
#define SEQ   2048
#define N1    2049
#define N1P   2064      // padded to 16-multiple
#define DIN   768
#define DIM   512
#define NH    8
#define DH    64
#define MLP   512
#define QKV3  1536
#define ROWS  (BATCH * N1P)   // 8256
#define RXIN  (BATCH * SEQ)   // 8192

// ---------------------------------------------------------------------------
// Tensor Data Mover support (gfx1250). Descriptor built per CDNA5 ISA D# spec.
// ---------------------------------------------------------------------------
#if defined(__AMDGCN__) && __has_builtin(__builtin_amdgcn_tensor_load_to_lds) && \
    __has_builtin(__builtin_amdgcn_s_wait_tensorcnt)
#define GEMM_TDM 1
#else
#define GEMM_TDM 0
#endif

#if GEMM_TDM
typedef unsigned int u32x4 __attribute__((ext_vector_type(4)));
typedef int          i32x4 __attribute__((ext_vector_type(4)));
typedef int          i32x8 __attribute__((ext_vector_type(8)));

// 2D tile load: tile = rows x 32 bf16 elements (contiguous K), row stride =
// stride_elems, into LDS at lds_off, rows*64 bytes contiguous.
__device__ __forceinline__ void tdm_load_2d(const u16b* gptr, unsigned lds_off,
                                            unsigned rows, unsigned stride_elems) {
  unsigned long long ga = (unsigned long long)(size_t)gptr;
  u32x4 g0;
  g0[0] = 1u;                                        // count=1 (valid), user mode
  g0[1] = lds_off;                                   // LDS byte address
  g0[2] = (unsigned)(ga & 0xFFFFFFFFu);              // global_addr[31:0]
  g0[3] = (unsigned)((ga >> 32) & 0x01FFFFFFu)       // global_addr[56:32]
        | (2u << 30);                                // type=2 ("image")
  const unsigned td0 = 32u;                          // dim0 = 32 elements
  i32x8 g1;
  g1[0] = (int)(1u << 16);                           // data_size=1 (2 bytes)
  g1[1] = (int)(td0 << 16);                          // tensor_dim0[15:0] @ 63:48
  g1[2] = (int)((rows & 0xFFFFu) << 16);             // dim0 hi=0, tensor_dim1 lo
  g1[3] = (int)(td0 << 16);                          // dim1 hi=0, tile_dim0=32
  g1[4] = (int)(rows & 0xFFFFu);                     // tile_dim1=rows, tile_dim2=0
  g1[5] = (int)stride_elems;                         // tensor_dim0_stride[31:0]
  g1[6] = 0;                                         // stride hi, dim1_stride lo
  g1[7] = 0;
  i32x4 z4 = {0, 0, 0, 0};
#if defined(__clang_major__) && __clang_major__ >= 23
  i32x8 z8 = {0, 0, 0, 0, 0, 0, 0, 0};
  __builtin_amdgcn_tensor_load_to_lds(g0, g1, z4, z4, z8, 0);
#else
  __builtin_amdgcn_tensor_load_to_lds(g0, g1, z4, z4, 0);
#endif
}
#endif  // GEMM_TDM

// ---------------------------------------------------------------------------
// Elementwise converters
// ---------------------------------------------------------------------------
__global__ void f32_to_bf16_kernel(const float* __restrict__ src, u16b* __restrict__ dst, int count) {
  int i = blockIdx.x * 256 + threadIdx.x;
  if (i < count) dst[i] = f2bf(src[i]);
}

// dst[N][K] = bf16(src[K][N])  (store W transposed so WMMA B-frags are contiguous)
__global__ void w_transpose_bf16_kernel(const float* __restrict__ src, u16b* __restrict__ dst, int K, int N) {
  int i = blockIdx.x * 256 + threadIdx.x;
  if (i >= K * N) return;
  int k = i / N, n = i - k * N;
  dst[(size_t)n * K + k] = f2bf(src[i]);
}

// cls token into pos 0 of each batch row block; zero the 15 pad rows
__global__ void clspad_kernel(float* __restrict__ xbuf, const float* __restrict__ cls) {
  int slot = blockIdx.x;          // 0 = cls, 1..15 = pad rows
  int b = blockIdx.y;
  int pos = (slot == 0) ? 0 : (N1 - 1 + slot);
  float* row = xbuf + ((size_t)b * N1P + pos) * DIM;
  for (int i = threadIdx.x; i < DIM; i += blockDim.x)
    row[i] = (slot == 0) ? cls[i] : 0.0f;
}

// ---------------------------------------------------------------------------
// LayerNorm (f32 in -> bf16 out), one wave per row
// ---------------------------------------------------------------------------
__global__ __launch_bounds__(256) void ln_bf16_kernel(const float* __restrict__ x,
                                                      const float* __restrict__ g,
                                                      const float* __restrict__ be,
                                                      u16b* __restrict__ out, int rows, int dim) {
  int wave = threadIdx.x >> 5, lane = threadIdx.x & 31;
  int row = blockIdx.x * 8 + wave;
  if (row >= rows) return;
  const float* xr = x + (size_t)row * dim;
  float s = 0.f, s2 = 0.f;
  for (int i = lane; i < dim; i += 32) { float v = xr[i]; s += v; s2 += v * v; }
  for (int o = 16; o > 0; o >>= 1) { s += __shfl_xor(s, o, 32); s2 += __shfl_xor(s2, o, 32); }
  float mean = s / dim;
  float var  = s2 / dim - mean * mean;
  float inv  = rsqrtf(var + 1e-5f);
  u16b* orow = out + (size_t)row * dim;
  for (int i = lane; i < dim; i += 32)
    orow[i] = f2bf((xr[i] - mean) * inv * g[i] + be[i]);
}

// ---------------------------------------------------------------------------
// Generic bf16 WMMA GEMM: C[M,N] = A[M,K] * Bt[N,K]^T, fp32 accumulate.
// Block = 256 threads = 8 waves. Block tile 64x256; wave tile 32x64 (8 WMMA/step).
// A (64x32) and B (256x32) K-slabs staged in LDS by the Tensor Data Mover,
// double-buffered; wave 0 issues TDM + s_wait_tensorcnt, barrier publishes.
// Epilogues: 0=fc(+bias,gelu,row-remap to xbuf)  1=qkv split(q*scale,k,vT)
//            2=+bias+residual -> f32             3=+bias,gelu -> bf16
// ---------------------------------------------------------------------------
struct GemmParams {
  const u16b* A;  int lda;
  const u16b* Bt; int ldb;
  int M, N, K;
  const float* bias;
  const float* residual; int ldr;
  float* outF; u16b* outB; int ldo;
  u16b *q, *k, *vT; float scale;
};

template <int EPI>
__global__ __launch_bounds__(256) void gemm_bf16_kernel(GemmParams p) {
  const int wave = threadIdx.x >> 5;
  const int lane = threadIdx.x & 31;
  const int half = lane >> 4;
  const int l16  = lane & 15;
  const int wr   = (wave & 1) * 32;         // wave row offset in 64-row block tile
  const int wc   = (wave >> 1) * 64;        // wave col offset in 256-col block tile
  const int rowB = blockIdx.x * 64;
  const int colB = blockIdx.y * 256;

  v8f c[2][4];
#pragma unroll
  for (int f = 0; f < 2; ++f)
#pragma unroll
    for (int t = 0; t < 4; ++t) c[f][t] = v8f{};

#if GEMM_TDM
  __shared__ __align__(16) u16b Asl[2][64 * 32];
  __shared__ __align__(16) u16b Bsl[2][256 * 32];
  const bool issuer = (threadIdx.x < 32);    // wave 0 drives the TDM
  if (issuer) {
    tdm_load_2d(p.A  + (size_t)rowB * p.lda, (unsigned)(size_t)&Asl[0][0], 64,  (unsigned)p.lda);
    tdm_load_2d(p.Bt + (size_t)colB * p.ldb, (unsigned)(size_t)&Bsl[0][0], 256, (unsigned)p.ldb);
  }
  for (int k0 = 0, it = 0; k0 < p.K; k0 += 32, ++it) {
    const int cur = it & 1;
    if (issuer) __builtin_amdgcn_s_wait_tensorcnt(0);
    __syncthreads();                         // slab `cur` ready; prev slab free
    if (issuer && (k0 + 32 < p.K)) {
      tdm_load_2d(p.A  + (size_t)rowB * p.lda + k0 + 32,
                  (unsigned)(size_t)&Asl[cur ^ 1][0], 64, (unsigned)p.lda);
      tdm_load_2d(p.Bt + (size_t)colB * p.ldb + k0 + 32,
                  (unsigned)(size_t)&Bsl[cur ^ 1][0], 256, (unsigned)p.ldb);
    }
    BFrag a[2];
#pragma unroll
    for (int f = 0; f < 2; ++f) {
      const u16b* ar = &Asl[cur][(wr + f * 16 + l16) * 32];
      a[f].q[0] = *(const uint4*)(ar + 16 * half);
      a[f].q[1] = *(const uint4*)(ar + 16 * half + 8);
    }
#pragma unroll
    for (int t = 0; t < 4; ++t) {
      const u16b* br = &Bsl[cur][(wc + t * 16 + l16) * 32 + 16 * half];
      BFrag b;
      b.q[0] = *(const uint4*)(br);
      b.q[1] = *(const uint4*)(br + 8);
#pragma unroll
      for (int f = 0; f < 2; ++f)
        c[f][t] = __builtin_amdgcn_wmma_f32_16x16x32_bf16(false, a[f].v, false, b.v,
                                                          (short)0, c[f][t], false, false);
    }
  }
#else
  // Fallback: direct global-memory fragments (same math, no TDM builtins).
  for (int k0 = 0; k0 < p.K; k0 += 32) {
    BFrag a[2];
#pragma unroll
    for (int f = 0; f < 2; ++f) {
      const u16b* ar = p.A + (size_t)(rowB + wr + f * 16 + l16) * p.lda + k0;
      a[f].q[0] = *(const uint4*)(ar + 16 * half);
      a[f].q[1] = *(const uint4*)(ar + 16 * half + 8);
    }
#pragma unroll
    for (int t = 0; t < 4; ++t) {
      const u16b* br = p.Bt + (size_t)(colB + wc + t * 16 + l16) * p.ldb + k0 + 16 * half;
      BFrag b;
      b.q[0] = *(const uint4*)(br);
      b.q[1] = *(const uint4*)(br + 8);
#pragma unroll
      for (int f = 0; f < 2; ++f)
        c[f][t] = __builtin_amdgcn_wmma_f32_16x16x32_bf16(false, a[f].v, false, b.v,
                                                          (short)0, c[f][t], false, false);
    }
  }
#endif

#pragma unroll
  for (int f = 0; f < 2; ++f) {
#pragma unroll
    for (int t = 0; t < 4; ++t) {
#pragma unroll
      for (int r = 0; r < 8; ++r) {
        int row = rowB + wr + f * 16 + 8 * half + r;
        int col = colB + wc + t * 16 + l16;
        float val = c[f][t][r];
        if (EPI == 0) {                               // fc: gelu(x@W+b) -> xbuf[pos+1]
          val = geluf(val + p.bias[col]);
          int b_ = row >> 11, pos = (row & 2047) + 1;
          p.outF[((size_t)b_ * N1P + pos) * p.ldo + col] = val;
        } else if (EPI == 1) {                        // qkv split
          int b_ = row / N1P, pos = row % N1P;
          int g_ = col >> 9, hh = (col >> 6) & 7, d = col & 63;
          size_t bh = (size_t)(b_ * NH + hh);
          if (g_ == 0)      p.q[(bh * N1P + pos) * DH + d] = f2bf(val * p.scale);
          else if (g_ == 1) p.k[(bh * N1P + pos) * DH + d] = f2bf(val);
          else              p.vT[(bh * DH + d) * (size_t)N1P + pos] = f2bf(val);
        } else if (EPI == 2) {                        // + bias + residual -> f32
          p.outF[(size_t)row * p.ldo + col] =
              val + p.bias[col] + p.residual[(size_t)row * p.ldr + col];
        } else {                                      // + bias, gelu -> bf16
          p.outB[(size_t)row * p.ldo + col] = f2bf(geluf(val + p.bias[col]));
        }
      }
    }
  }
}

// ---------------------------------------------------------------------------
// Flash attention: one wave per 16-row query tile, stream keys in chunks of 32.
// q pre-scaled by 1/sqrt(DH). Mask matches reference: only (q pad & k pad).
// ---------------------------------------------------------------------------
__global__ __launch_bounds__(256) void attn_kernel(const u16b* __restrict__ Q,
                                                   const u16b* __restrict__ K,
                                                   const u16b* __restrict__ VT,
                                                   const int* __restrict__ lens,
                                                   u16b* __restrict__ O) {
  __shared__ __align__(16) u16b Plds[8][16][32];
  const int wave = threadIdx.x >> 5;
  const int lane = threadIdx.x & 31;
  const int half = lane >> 4;
  const int l16  = lane & 15;
  const int bh = blockIdx.x;
  const int b_ = bh >> 3, h_ = bh & 7;
  const int qt = blockIdx.y * 8 + wave;
  const int qrow0 = qt * 16;
  if (qrow0 >= N1P) return;
  const int len1 = lens[b_] + 1;

  const u16b* qb = Q  + (size_t)bh * N1P * DH;
  const u16b* kb = K  + (size_t)bh * N1P * DH;
  const u16b* vb = VT + (size_t)bh * DH * N1P;

  BFrag qa0, qa1;                                      // q tile, K=0..31 / 32..63
  {
    const u16b* qrow = qb + (size_t)(qrow0 + l16) * DH;
    qa0.q[0] = *(const uint4*)(qrow + 16 * half);
    qa0.q[1] = *(const uint4*)(qrow + 16 * half + 8);
    qa1.q[0] = *(const uint4*)(qrow + 32 + 16 * half);
    qa1.q[1] = *(const uint4*)(qrow + 32 + 16 * half + 8);
  }

  v8f oacc[4] = {v8f{}, v8f{}, v8f{}, v8f{}};          // 16 x 64 output
  float m[8], lsum[8];
#pragma unroll
  for (int r = 0; r < 8; ++r) { m[r] = -3.0e38f; lsum[r] = 0.0f; }

  for (int kc = 0; kc < N1P; kc += 32) {
    v8f S[2];
#pragma unroll
    for (int j = 0; j < 2; ++j) {                      // two 16x16 score tiles
      const int kcol0 = kc + 16 * j;
      const u16b* krow = kb + (size_t)(kcol0 + l16) * DH;
      BFrag kf;
      v8f acc = v8f{};
      kf.q[0] = *(const uint4*)(krow + 16 * half);
      kf.q[1] = *(const uint4*)(krow + 16 * half + 8);
      acc = __builtin_amdgcn_wmma_f32_16x16x32_bf16(false, qa0.v, false, kf.v,
                                                    (short)0, acc, false, false);
      kf.q[0] = *(const uint4*)(krow + 32 + 16 * half);
      kf.q[1] = *(const uint4*)(krow + 32 + 16 * half + 8);
      acc = __builtin_amdgcn_wmma_f32_16x16x32_bf16(false, qa1.v, false, kf.v,
                                                    (short)0, acc, false, false);
      S[j] = acc;
    }
    // mask (q already carries the 1/sqrt(DH) scale)
#pragma unroll
    for (int j = 0; j < 2; ++j)
#pragma unroll
      for (int r = 0; r < 8; ++r) {
        int col  = kc + 16 * j + l16;
        int qpos = qrow0 + 8 * half + r;
        float s = S[j][r];
        if (col >= N1 || (qpos >= len1 && col >= len1)) s = -3.0e38f;
        S[j][r] = s;
      }
    // online softmax; row stats reduced across the 16-lane half-wave
    float alpha[8];
#pragma unroll
    for (int r = 0; r < 8; ++r) {
      float mx = fmaxf(S[0][r], S[1][r]);
#pragma unroll
      for (int w = 8; w > 0; w >>= 1) mx = fmaxf(mx, __shfl_xor(mx, w, 16));
      float mn = fmaxf(m[r], mx);
      float p0 = __expf(S[0][r] - mn), p1 = __expf(S[1][r] - mn);
      S[0][r] = p0; S[1][r] = p1;
      float su = p0 + p1;
#pragma unroll
      for (int w = 8; w > 0; w >>= 1) su += __shfl_xor(su, w, 16);
      alpha[r] = __expf(m[r] - mn);
      lsum[r]  = lsum[r] * alpha[r] + su;
      m[r]     = mn;
    }
#pragma unroll
    for (int r = 0; r < 8; ++r) {
      oacc[0][r] *= alpha[r]; oacc[1][r] *= alpha[r];
      oacc[2][r] *= alpha[r]; oacc[3][r] *= alpha[r];
    }
    // stage P (C-layout) through LDS and reload as an A fragment
#pragma unroll
    for (int j = 0; j < 2; ++j)
#pragma unroll
      for (int r = 0; r < 8; ++r)
        Plds[wave][8 * half + r][16 * j + l16] = f2bf(S[j][r]);
    BFrag pa;
    {
      const u16b* prow = &Plds[wave][l16][0];
      pa.q[0] = *(const uint4*)(prow + 16 * half);
      pa.q[1] = *(const uint4*)(prow + 16 * half + 8);
    }
    // O += P @ V   (vT rows = head dims, contiguous in key index)
#pragma unroll
    for (int t = 0; t < 4; ++t) {
      const u16b* vrow = vb + (size_t)(t * 16 + l16) * N1P + kc + 16 * half;
      BFrag vf;
      vf.q[0] = *(const uint4*)(vrow);
      vf.q[1] = *(const uint4*)(vrow + 8);
      oacc[t] = __builtin_amdgcn_wmma_f32_16x16x32_bf16(false, pa.v, false, vf.v,
                                                        (short)0, oacc[t], false, false);
    }
  }
  // finalize and scatter into the concatenated-head output buffer (bf16)
#pragma unroll
  for (int t = 0; t < 4; ++t)
#pragma unroll
    for (int r = 0; r < 8; ++r) {
      int row = qrow0 + 8 * half + r;
      float val = oacc[t][r] / lsum[r];
      O[((size_t)b_ * N1P + row) * (NH * DH) + h_ * DH + t * 16 + l16] = f2bf(val);
    }
}

// ---------------------------------------------------------------------------
// Final LN(row 0) + head: one wave per batch
// ---------------------------------------------------------------------------
__global__ void head_kernel(const float* __restrict__ xbuf, const float* __restrict__ g,
                            const float* __restrict__ be, const float* __restrict__ hw,
                            const float* __restrict__ hb, float* __restrict__ out) {
  int b_ = blockIdx.x, lane = threadIdx.x;
  const float* xr = xbuf + (size_t)b_ * N1P * DIM;     // cls row (pos 0)
  float s = 0.f, s2 = 0.f;
  for (int i = lane; i < DIM; i += 32) { float v = xr[i]; s += v; s2 += v * v; }
  for (int o = 16; o > 0; o >>= 1) { s += __shfl_xor(s, o, 32); s2 += __shfl_xor(s2, o, 32); }
  float mean = s / DIM, var = s2 / DIM - mean * mean, inv = rsqrtf(var + 1e-5f);
  float d0 = 0.f, d1 = 0.f;
  for (int i = lane; i < DIM; i += 32) {
    float xn = (xr[i] - mean) * inv * g[i] + be[i];
    d0 += xn * hw[i * 2 + 0];
    d1 += xn * hw[i * 2 + 1];
  }
  for (int o = 16; o > 0; o >>= 1) { d0 += __shfl_xor(d0, o, 32); d1 += __shfl_xor(d1, o, 32); }
  if (lane == 0) { out[b_ * 2 + 0] = d0 + hb[0]; out[b_ * 2 + 1] = d1 + hb[1]; }
}

// ---------------------------------------------------------------------------
extern "C" void kernel_launch(void* const* d_in, const int* in_sizes, int n_in,
                              void* d_out, int out_size, void* d_ws, size_t ws_size,
                              hipStream_t stream) {
  (void)in_sizes; (void)n_in; (void)out_size; (void)ws_size;
  const float* x       = (const float*)d_in[0];
  const int*   lens    = (const int*)  d_in[1];
  const float* cls     = (const float*)d_in[2];
  const float* fc_w    = (const float*)d_in[3];
  const float* fc_b    = (const float*)d_in[4];
  const float* ln1_g   = (const float*)d_in[5];
  const float* ln1_b   = (const float*)d_in[6];
  const float* qkv_w   = (const float*)d_in[7];
  const float* out_w   = (const float*)d_in[8];
  const float* out_b   = (const float*)d_in[9];
  const float* ln2_g   = (const float*)d_in[10];
  const float* ln2_b   = (const float*)d_in[11];
  const float* ff1_w   = (const float*)d_in[12];
  const float* ff1_b   = (const float*)d_in[13];
  const float* ff2_w   = (const float*)d_in[14];
  const float* ff2_b   = (const float*)d_in[15];
  const float* lnf_g   = (const float*)d_in[16];
  const float* lnf_b   = (const float*)d_in[17];
  const float* head_w  = (const float*)d_in[18];
  const float* head_b  = (const float*)d_in[19];
  float* out = (float*)d_out;

  // bump allocator over workspace
  char* ws = (char*)d_ws; size_t off = 0;
  auto alloc = [&](size_t bytes) -> void* {
    void* p = ws + off;
    off += (bytes + 255) & ~(size_t)255;
    return p;
  };
  u16b* xb16 = (u16b*)alloc((size_t)RXIN * DIN * 2);
  u16b* wfc  = (u16b*)alloc((size_t)DIM * DIN * 2);
  u16b* wqkv[2], *wout[2], *wf1[2], *wf2[2];
  for (int l = 0; l < 2; ++l) {
    wqkv[l] = (u16b*)alloc((size_t)QKV3 * DIM * 2);
    wout[l] = (u16b*)alloc((size_t)DIM * DIM * 2);
    wf1[l]  = (u16b*)alloc((size_t)MLP * DIM * 2);
    wf2[l]  = (u16b*)alloc((size_t)DIM * MLP * 2);
  }
  float* xbuf = (float*)alloc((size_t)ROWS * DIM * 4);   // residual stream (f32)
  u16b*  xn   = (u16b*) alloc((size_t)ROWS * DIM * 2);   // LN output (bf16)
  u16b*  qB   = (u16b*) alloc((size_t)BATCH * NH * N1P * DH * 2);
  u16b*  kB   = (u16b*) alloc((size_t)BATCH * NH * N1P * DH * 2);
  u16b*  vT   = (u16b*) alloc((size_t)BATCH * NH * DH * N1P * 2);
  u16b*  ob   = (u16b*) alloc((size_t)ROWS * (NH * DH) * 2);
  u16b*  hff  = (u16b*) alloc((size_t)ROWS * MLP * 2);

  // --- prep: bf16 activations + transposed bf16 weights ---
  f32_to_bf16_kernel<<<(RXIN * DIN + 255) / 256, 256, 0, stream>>>(x, xb16, RXIN * DIN);
  w_transpose_bf16_kernel<<<(DIN * DIM + 255) / 256, 256, 0, stream>>>(fc_w, wfc, DIN, DIM);
  for (int l = 0; l < 2; ++l) {
    w_transpose_bf16_kernel<<<(DIM * QKV3 + 255) / 256, 256, 0, stream>>>(qkv_w + (size_t)l * DIM * QKV3, wqkv[l], DIM, QKV3);
    w_transpose_bf16_kernel<<<(DIM * DIM + 255) / 256, 256, 0, stream>>>(out_w + (size_t)l * DIM * DIM, wout[l], DIM, DIM);
    w_transpose_bf16_kernel<<<(DIM * MLP + 255) / 256, 256, 0, stream>>>(ff1_w + (size_t)l * DIM * MLP, wf1[l], DIM, MLP);
    w_transpose_bf16_kernel<<<(MLP * DIM + 255) / 256, 256, 0, stream>>>(ff2_w + (size_t)l * MLP * DIM, wf2[l], MLP, DIM);
  }

  // --- fc + GELU -> xbuf rows 1..2048 per batch ---
  {
    GemmParams p = {};
    p.A = xb16; p.lda = DIN; p.Bt = wfc; p.ldb = DIN;
    p.M = RXIN; p.N = DIM; p.K = DIN;
    p.bias = fc_b; p.outF = xbuf; p.ldo = DIM;
    gemm_bf16_kernel<0><<<dim3(RXIN / 64, DIM / 256), 256, 0, stream>>>(p);
  }
  clspad_kernel<<<dim3(16, BATCH), 256, 0, stream>>>(xbuf, cls);

  const int attnGY = ((N1P / 16) + 7) / 8;   // query-tile groups
  for (int l = 0; l < 2; ++l) {
    // LN1
    ln_bf16_kernel<<<ROWS / 8, 256, 0, stream>>>(xbuf, ln1_g + l * DIM, ln1_b + l * DIM, xn, ROWS, DIM);
    // QKV projection + split
    {
      GemmParams p = {};
      p.A = xn; p.lda = DIM; p.Bt = wqkv[l]; p.ldb = DIM;
      p.M = ROWS; p.N = QKV3; p.K = DIM;
      p.q = qB; p.k = kB; p.vT = vT; p.scale = 0.125f;  // DH^-0.5
      gemm_bf16_kernel<1><<<dim3(ROWS / 64, QKV3 / 256), 256, 0, stream>>>(p);
    }
    // Attention
    attn_kernel<<<dim3(BATCH * NH, attnGY), 256, 0, stream>>>(qB, kB, vT, lens, ob);
    // Output projection + bias + residual
    {
      GemmParams p = {};
      p.A = ob; p.lda = NH * DH; p.Bt = wout[l]; p.ldb = NH * DH;
      p.M = ROWS; p.N = DIM; p.K = NH * DH;
      p.bias = out_b + l * DIM; p.residual = xbuf; p.ldr = DIM;
      p.outF = xbuf; p.ldo = DIM;
      gemm_bf16_kernel<2><<<dim3(ROWS / 64, DIM / 256), 256, 0, stream>>>(p);
    }
    // LN2
    ln_bf16_kernel<<<ROWS / 8, 256, 0, stream>>>(xbuf, ln2_g + l * DIM, ln2_b + l * DIM, xn, ROWS, DIM);
    // FF1 + GELU
    {
      GemmParams p = {};
      p.A = xn; p.lda = DIM; p.Bt = wf1[l]; p.ldb = DIM;
      p.M = ROWS; p.N = MLP; p.K = DIM;
      p.bias = ff1_b + l * MLP; p.outB = hff; p.ldo = MLP;
      gemm_bf16_kernel<3><<<dim3(ROWS / 64, MLP / 256), 256, 0, stream>>>(p);
    }
    // FF2 + bias + residual
    {
      GemmParams p = {};
      p.A = hff; p.lda = MLP; p.Bt = wf2[l]; p.ldb = MLP;
      p.M = ROWS; p.N = DIM; p.K = MLP;
      p.bias = ff2_b + l * DIM; p.residual = xbuf; p.ldr = DIM;
      p.outF = xbuf; p.ldo = DIM;
      gemm_bf16_kernel<2><<<dim3(ROWS / 64, DIM / 256), 256, 0, stream>>>(p);
    }
  }
  // Final LN + classification head on the cls token
  head_kernel<<<BATCH, 32, 0, stream>>>(xbuf, lnf_g, lnf_b, head_w, head_b, out);
}
